// IterativeGaussianProcess_8744553415327
// MI455X (gfx1250) — matrix operand order repaired and checked by reading, not compiled, not run on hardware
//
#include <hip/hip_runtime.h>
#include <cstdint>
#include <cstddef>

// ---------------------------------------------------------------------------
// Problem constants (match reference: n=8192, d=128, m=16, 64 CG iterations)
// ---------------------------------------------------------------------------
#define N_PTS 8192
#define DIMS  128
#define NCOL  32        // padded RHS columns (17 used, rest exact zero)
#define NOUT  17
#define CG_ITERS 64
#define EPS_CG 1e-10f

typedef _Float16 v8h  __attribute__((ext_vector_type(8)));
typedef _Float16 v16h __attribute__((ext_vector_type(16)));
typedef float    v8f  __attribute__((ext_vector_type(8)));

// ---------------------------------------------------------------------------
// Workspace layout (byte offsets from d_ws). Total ~141 MB.
// ---------------------------------------------------------------------------
static constexpr size_t OFF_GAMMA = 0;                          // 32 f32
static constexpr size_t OFF_ALPHA = 128;                        // 32 f32
static constexpr size_t OFF_BETA  = 256;                        // 32 f32
static constexpr size_t OFF_RHS   = 384;                        // 32 f32
static constexpr size_t OFF_SQ    = 1024;                       // 8192 f32
static constexpr size_t OFF_XH    = 40960;                      // X in f16: 8192*128*2 = 2 MB
static constexpr size_t OFF_X     = OFF_XH + (size_t)N_PTS * DIMS * 2;   // x  f32 [8192][32]
static constexpr size_t OFF_R     = OFF_X  + (size_t)N_PTS * NCOL * 4;   // r  f32
static constexpr size_t OFF_P     = OFF_R  + (size_t)N_PTS * NCOL * 4;   // p  f32
static constexpr size_t OFF_AP    = OFF_P  + (size_t)N_PTS * NCOL * 4;   // Ap f32
static constexpr size_t OFF_PT    = OFF_AP + (size_t)N_PTS * NCOL * 4;   // pT f16 [32][8192]
static constexpr size_t OFF_KH    = OFF_PT + (size_t)NCOL * N_PTS * 2;   // K  f16 [8192][8192] = 128 MB

#define WS_F(off) ((float*)(ws + (off)))
#define WS_H(off) ((_Float16*)(ws + (off)))

// ---------------------------------------------------------------------------
// WMMA fragment loaders (CDNA5 wave32 VGPR layouts, 16-bit operands)
// ---------------------------------------------------------------------------
// A matrix 16x32 (MxK), row-major source with leading dim `ld`:
//   lane 0..15 : row M=lane,   halves = K[0..7]  then K[16..23]
//   lane 16..31: row M=lane-16, halves = K[8..15] then K[24..31]
__device__ __forceinline__ v16h load_frag_a(const _Float16* src, int ld,
                                            int row0, int k0, int lane) {
  int m  = lane & 15;
  int hk = lane >> 4;
  const _Float16* b = src + (size_t)(row0 + m) * ld + k0 + hk * 8;
  v8h lo = *(const v8h*)(b);
  v8h hi = *(const v8h*)(b + 16);
  return __builtin_shufflevector(lo, hi, 0,1,2,3,4,5,6,7,8,9,10,11,12,13,14,15);
}

// B matrix 32x16 (KxN) from a TRANSPOSED source srcT[N][ld] (row n = column n):
//   lane 0..15 : col N=lane,    halves = K[0..7]  then K[16..23]
//   lane 16..31: col N=lane-16, halves = K[8..15] then K[24..31]
__device__ __forceinline__ v16h load_frag_b(const _Float16* srcT, int ld,
                                            int n0, int k0, int lane) {
  int n  = lane & 15;
  int hk = lane >> 4;
  const _Float16* b = srcT + (size_t)(n0 + n) * ld + k0 + hk * 8;
  v8h lo = *(const v8h*)(b);
  v8h hi = *(const v8h*)(b + 16);
  return __builtin_shufflevector(lo, hi, 0,1,2,3,4,5,6,7,8,9,10,11,12,13,14,15);
}

__device__ __forceinline__ float block_reduce_256(float v, float* red) {
  int t = threadIdx.x;
  red[t] = v;
  __syncthreads();
  for (int s = 128; s > 0; s >>= 1) {
    if (t < s) red[t] += red[t + s];
    __syncthreads();
  }
  return red[0];
}

// ---------------------------------------------------------------------------
// Kernel 1: convert X -> f16, compute per-row squared norms. 1 block per row.
// ---------------------------------------------------------------------------
__global__ void __launch_bounds__(128) gp_prep_x(const float* __restrict__ X, char* ws) {
  __shared__ float red[128];
  int i = blockIdx.x, t = threadIdx.x;
  float v = X[(size_t)i * DIMS + t];
  WS_H(OFF_XH)[(size_t)i * DIMS + t] = (_Float16)v;
  red[t] = v * v;
  __syncthreads();
  for (int s = 64; s > 0; s >>= 1) {
    if (t < s) red[t] += red[t + s];
    __syncthreads();
  }
  if (t == 0) WS_F(OFF_SQ)[i] = red[0];
}

// ---------------------------------------------------------------------------
// Kernel 2: build K (f16) = outputscale * exp(-0.5*d2/l^2) via f16 WMMA SYRK.
// grid (64, 512), 256 threads (8 waves). Each wave -> one 16x16 tile.
// ---------------------------------------------------------------------------
__global__ void __launch_bounds__(256) gp_build_k(char* ws,
                                                  const float* __restrict__ ls,
                                                  const float* __restrict__ os) {
  const _Float16* xh = WS_H(OFF_XH);
  const float*    sq = WS_F(OFF_SQ);
  _Float16*       kh = WS_H(OFF_KH);

  int wave = threadIdx.x >> 5, lane = threadIdx.x & 31;
  int j0 = (blockIdx.x * 8 + wave) * 16;   // column tile
  int i0 = blockIdx.y * 16;                // row tile

  v8f acc = {};
#pragma unroll
  for (int kk = 0; kk < DIMS; kk += 32) {
    v16h a = load_frag_a(xh, DIMS, i0, kk, lane);   // rows of X
    v16h b = load_frag_b(xh, DIMS, j0, kk, lane);   // rows of X as B columns (X^T)
    acc = __builtin_amdgcn_wmma_f32_16x16x32_f16(false, a, false, b,
                                                 (short)0, acc, false, false);
  }

  float l    = ls[0];
  float inv2 = 0.5f / (l * l);
  float osv  = os[0];
  int hk = lane >> 4, col = lane & 15;
  int j  = j0 + col;
  float sqj = sq[j];
#pragma unroll
  for (int v = 0; v < 8; ++v) {
    int i = i0 + v + hk * 8;
    float d2 = sq[i] + sqj - 2.0f * acc[v];
    d2 = fmaxf(d2, 0.0f);
    kh[(size_t)i * N_PTS + j] = (_Float16)(osv * __expf(-d2 * inv2));
  }
}

// ---------------------------------------------------------------------------
// Kernel 3: build b = [y, normalized probes], rhs norms, CG init state.
// One block per padded column (32 blocks).
// ---------------------------------------------------------------------------
__global__ void __launch_bounds__(256) gp_setup(const float* __restrict__ y,
                                                const float* __restrict__ probes,
                                                char* ws) {
  __shared__ float red[256];
  int c = blockIdx.x, t = threadIdx.x;

  float ss = 0.f;
  for (int i = t; i < N_PTS; i += 256) {
    float s = (c == 0) ? y[i] : ((c <= 16) ? probes[(size_t)i * 16 + (c - 1)] : 0.f);
    ss += s * s;
  }
  ss = block_reduce_256(ss, red);

  float scale, rhsg;
  if (c == 0) {
    float rn = sqrtf(ss);               // rhs_norm of y column
    rhsg  = (rn < EPS_CG) ? 1.f : rn;
    scale = 1.f / rhsg;
  } else {
    float nb    = sqrtf(ss);            // probe column norm
    float denom = nb + EPS_CG;          // pn = probes/denom
    float rhs   = nb / denom;           // ||pn||
    rhsg  = (rhs < EPS_CG) ? 1.f : rhs;
    scale = 1.f / (denom * rhsg);       // bn = probes * scale
  }
  float gam = ss * scale * scale;       // gamma0 = sum bn^2

  float* x = WS_F(OFF_X);
  float* r = WS_F(OFF_R);
  float* p = WS_F(OFF_P);
  _Float16* pT = WS_H(OFF_PT);
  for (int i = t; i < N_PTS; i += 256) {
    float s = (c == 0) ? y[i] : ((c <= 16) ? probes[(size_t)i * 16 + (c - 1)] : 0.f);
    float bn = s * scale;
    size_t o = (size_t)i * NCOL + c;
    x[o] = 0.f;
    r[o] = bn;
    p[o] = bn;
    pT[(size_t)c * N_PTS + i] = (_Float16)bn;
  }
  if (t == 0) {
    WS_F(OFF_GAMMA)[c] = gam;
    WS_F(OFF_RHS)[c]   = rhsg;
  }
}

// ---------------------------------------------------------------------------
// Kernel 4: Ap = K @ p + sigma^2 * p   (the CG matvec; K is L2-resident f16)
// 64 blocks x 256 threads; each wave owns one 16-row strip x 32 cols.
// ---------------------------------------------------------------------------
__global__ void __launch_bounds__(256) gp_gemm_ap(char* ws, const float* __restrict__ nu) {
  const _Float16* kh = WS_H(OFF_KH);
  const _Float16* pT = WS_H(OFF_PT);
  const float*    p  = WS_F(OFF_P);
  float*          Ap = WS_F(OFF_AP);

  int wave = threadIdx.x >> 5, lane = threadIdx.x & 31;
  int row0 = (blockIdx.x * 8 + wave) * 16;

  v8f acc0 = {};
  v8f acc1 = {};
  const _Float16* arow = kh + (size_t)(row0 + (lane & 15)) * N_PTS;
  for (int kk = 0; kk < N_PTS; kk += 32) {
    v16h a  = load_frag_a(kh, N_PTS, row0, kk, lane);
    v16h b0 = load_frag_b(pT, N_PTS, 0,  kk, lane);
    v16h b1 = load_frag_b(pT, N_PTS, 16, kk, lane);
    acc0 = __builtin_amdgcn_wmma_f32_16x16x32_f16(false, a, false, b0,
                                                  (short)0, acc0, false, false);
    acc1 = __builtin_amdgcn_wmma_f32_16x16x32_f16(false, a, false, b1,
                                                  (short)0, acc1, false, false);
    __builtin_prefetch(arow + kk + 256, 0, 0);   // global_prefetch_b8 ahead on K row
  }

  // sigma = 1e-3 + softplus(noise_u); s2 = sigma^2
  float u  = nu[0];
  float sp = (u > 20.f) ? u : log1pf(__expf(u));
  float sg = 1e-3f + sp;
  float s2 = sg * sg;

  int hk = lane >> 4, col = lane & 15;
#pragma unroll
  for (int v = 0; v < 8; ++v) {
    size_t row = (size_t)(row0 + v + hk * 8);
    size_t o0 = row * NCOL + col;
    size_t o1 = o0 + 16;
    Ap[o0] = acc0[v] + s2 * p[o0];
    Ap[o1] = acc1[v] + s2 * p[o1];
  }
}

// ---------------------------------------------------------------------------
// Kernel 5: alpha[c] = gamma[c] / (sum_i p*Ap + eps).  One block per column.
// ---------------------------------------------------------------------------
__global__ void __launch_bounds__(256) gp_alpha(char* ws) {
  __shared__ float red[256];
  int c = blockIdx.x, t = threadIdx.x;
  const float* p  = WS_F(OFF_P);
  const float* Ap = WS_F(OFF_AP);
  float s = 0.f;
  for (int i = t; i < N_PTS; i += 256) {
    size_t o = (size_t)i * NCOL + c;
    s += p[o] * Ap[o];
  }
  s = block_reduce_256(s, red);
  if (t == 0) WS_F(OFF_ALPHA)[c] = WS_F(OFF_GAMMA)[c] / (s + EPS_CG);
}

// ---------------------------------------------------------------------------
// Kernel 6: x += alpha*p ; r -= alpha*Ap   (elementwise, 8192x32)
// ---------------------------------------------------------------------------
__global__ void __launch_bounds__(256) gp_update_xr(char* ws) {
  size_t idx = (size_t)blockIdx.x * 256 + threadIdx.x;
  int c = (int)(idx & (NCOL - 1));
  float a = WS_F(OFF_ALPHA)[c];
  float* x  = WS_F(OFF_X);
  float* r  = WS_F(OFF_R);
  const float* p  = WS_F(OFF_P);
  const float* Ap = WS_F(OFF_AP);
  x[idx] += a * p[idx];
  r[idx] -= a * Ap[idx];
}

// ---------------------------------------------------------------------------
// Kernel 7: gamma_new = sum r^2 ; beta = gamma_new/(gamma+eps) ; gamma <- new
// ---------------------------------------------------------------------------
__global__ void __launch_bounds__(256) gp_beta(char* ws) {
  __shared__ float red[256];
  int c = blockIdx.x, t = threadIdx.x;
  const float* r = WS_F(OFF_R);
  float s = 0.f;
  for (int i = t; i < N_PTS; i += 256) {
    float rv = r[(size_t)i * NCOL + c];
    s += rv * rv;
  }
  s = block_reduce_256(s, red);
  if (t == 0) {
    WS_F(OFF_BETA)[c]  = s / (WS_F(OFF_GAMMA)[c] + EPS_CG);
    WS_F(OFF_GAMMA)[c] = s;
  }
}

// ---------------------------------------------------------------------------
// Kernel 8: p = r + beta*p  (f32) and refresh transposed f16 copy for WMMA B.
// ---------------------------------------------------------------------------
__global__ void __launch_bounds__(256) gp_update_p(char* ws) {
  size_t idx = (size_t)blockIdx.x * 256 + threadIdx.x;
  int c = (int)(idx & (NCOL - 1));
  size_t row = idx >> 5;
  float b = WS_F(OFF_BETA)[c];
  float* p = WS_F(OFF_P);
  const float* r = WS_F(OFF_R);
  float np = r[idx] + b * p[idx];
  p[idx] = np;
  WS_H(OFF_PT)[(size_t)c * N_PTS + row] = (_Float16)np;
}

// ---------------------------------------------------------------------------
// Kernel 9: solution = x * rhs_norm, packed as [8192,17] f32.
// ---------------------------------------------------------------------------
__global__ void __launch_bounds__(256) gp_finalize(char* ws, float* __restrict__ out) {
  int idx = blockIdx.x * 256 + threadIdx.x;
  if (idx >= N_PTS * NOUT) return;
  int i = idx / NOUT;
  int c = idx - i * NOUT;
  out[idx] = WS_F(OFF_X)[(size_t)i * NCOL + c] * WS_F(OFF_RHS)[c];
}

// ---------------------------------------------------------------------------
// Host launcher
// ---------------------------------------------------------------------------
extern "C" void kernel_launch(void* const* d_in, const int* in_sizes, int n_in,
                              void* d_out, int out_size, void* d_ws, size_t ws_size,
                              hipStream_t stream) {
  const float* X      = (const float*)d_in[0];   // [8192,128]
  const float* y      = (const float*)d_in[1];   // [8192]
  const float* probes = (const float*)d_in[2];   // [8192,16]
  const float* ls     = (const float*)d_in[3];   // scalar
  const float* os     = (const float*)d_in[4];   // scalar
  const float* nu     = (const float*)d_in[5];   // scalar
  char* ws   = (char*)d_ws;
  float* out = (float*)d_out;

  // Stage 1: X -> f16 + row squared norms
  gp_prep_x<<<N_PTS, 128, 0, stream>>>(X, ws);

  // Stage 2: K (f16, 128 MB -> L2 resident) via WMMA SYRK + RBF transform
  gp_build_k<<<dim3(64, N_PTS / 16), 256, 0, stream>>>(ws, ls, os);

  // Stage 3: CG init (b, rhs norms, x0/r0/p0/gamma0, transposed f16 p)
  gp_setup<<<NCOL, 256, 0, stream>>>(y, probes, ws);

  // Stage 4: 64 fixed CG iterations
  for (int it = 0; it < CG_ITERS; ++it) {
    gp_gemm_ap<<<N_PTS / (16 * 8), 256, 0, stream>>>(ws, nu);   // Ap = K@p + s2*p
    gp_alpha<<<NCOL, 256, 0, stream>>>(ws);                     // alpha
    gp_update_xr<<<(N_PTS * NCOL) / 256, 256, 0, stream>>>(ws); // x, r
    gp_beta<<<NCOL, 256, 0, stream>>>(ws);                      // beta, gamma
    gp_update_p<<<(N_PTS * NCOL) / 256, 256, 0, stream>>>(ws);  // p, pT(f16)
  }

  // Stage 5: scale by rhs_norm and emit [8192,17]
  gp_finalize<<<(N_PTS * NOUT + 255) / 256, 256, 0, stream>>>(ws, out);
}
